// EGNN_38963943309718
// MI455X (gfx1250) — compile-verified, matrix operand order checked by reading, not run on hardware
//
#include <hip/hip_runtime.h>
#include <hip/hip_bf16.h>
#include <math.h>

// ---------------- problem constants ----------------
#define Bn 2
#define Nn 512
#define HD 64
#define U  128
#define KE 160          // padded K for edge feature GEMM (129 -> 160)
#define KH 192          // K for node GEMM (U + HD)

typedef __attribute__((ext_vector_type(16))) _Float16 v16h;
typedef __attribute__((ext_vector_type(8)))  float    v8f;

// ---------------- workspace layout (bytes) ----------------
#define OFF_MI     0u          // f32 [Bn*Nn][U]          = 524288
#define OFF_WE1T   524288u     // f16 [U][KE]             =  40960
#define OFF_WE2T   565248u     // f16 [U][U]              =  32768
#define OFF_WX1T   598016u     // f16 [U][U]              =  32768
#define OFF_WX2T   630784u     // f16 [U][U]              =  32768
#define OFF_WH1T   663552u     // f16 [U][KH]             =  49152
#define OFF_WH2OT  712704u     // f16 [HD][U] (Wh2@Wout)^T=  16384
#define OFF_WINF   729088u     // f32 [U]
#define OFF_WX3    729600u     // f32 [U]
#define OFF_BH2O   730112u     // f32 [HD]

// ---------------- helpers ----------------
__device__ __forceinline__ float eluf(float x) {
  return x > 0.0f ? x : (__expf(x) - 1.0f);
}
__device__ __forceinline__ float sigf(float x) {
  return 1.0f / (1.0f + __expf(-x));
}

// WMMA f32 += f16[16x32] * f16[32x16]
__device__ __forceinline__ v8f wmma32(v16h a, v16h b, v8f c) {
  return __builtin_amdgcn_wmma_f32_16x16x32_f16(false, a, false, b, (short)0, c,
                                                false, false);
}

// A-fragment gather from row-major f16 [16][ldK] in LDS.
// 16-bit A 16x32 layout: lanes 0-15 -> row=lane, K = kb+{0..7, 16..23};
// lanes 16-31 -> row=lane-16, K = kb+{8..15, 24..31}.
__device__ __forceinline__ v16h ldsA(const _Float16* __restrict__ base, int ldK,
                                     int kb, int lane) {
  const _Float16* p = base + (lane & 15) * ldK + kb + ((lane & 16) ? 8 : 0);
  v16h a;
#pragma unroll
  for (int t = 0; t < 8; ++t) { a[t] = p[t]; a[t + 8] = p[16 + t]; }
  return a;
}

// B-fragment gather from N-major f16 weights WT[N][K] in global memory
// (same gather pattern as A, with "row" = output column).
__device__ __forceinline__ v16h glbB(const _Float16* __restrict__ wt, int ldK,
                                     int ncolBase, int kb, int lane) {
  const _Float16* p =
      wt + (ncolBase + (lane & 15)) * ldK + kb + ((lane & 16) ? 8 : 0);
  v16h b;
#pragma unroll
  for (int t = 0; t < 8; ++t) { b[t] = p[t]; b[t + 8] = p[16 + t]; }
  return b;
}

// ---------------- prep: repack weights to f16 transposed ----------------
__global__ void prep_pack(const float* __restrict__ We1,
                          const float* __restrict__ We2,
                          const float* __restrict__ Wx1,
                          const float* __restrict__ Wx2,
                          const float* __restrict__ Wh1,
                          const float* __restrict__ Winf,
                          const float* __restrict__ Wx3,
                          _Float16* We1T, _Float16* We2T, _Float16* Wx1T,
                          _Float16* Wx2T, _Float16* Wh1T, float* WinfF,
                          float* Wx3F) {
  int idx = blockIdx.x * blockDim.x + threadIdx.x;
  if (idx < U * KE) {  // We1T[n][k] = (k<129) ? We1[k][n] : 0
    int n = idx / KE, k = idx - n * KE;
    We1T[idx] = (k < 1 + 2 * HD) ? (_Float16)We1[k * U + n] : (_Float16)0.0f;
    return;
  }
  idx -= U * KE;
  if (idx < U * U) { int n = idx >> 7, k = idx & 127; We2T[idx] = (_Float16)We2[k * U + n]; return; }
  idx -= U * U;
  if (idx < U * U) { int n = idx >> 7, k = idx & 127; Wx1T[idx] = (_Float16)Wx1[k * U + n]; return; }
  idx -= U * U;
  if (idx < U * U) { int n = idx >> 7, k = idx & 127; Wx2T[idx] = (_Float16)Wx2[k * U + n]; return; }
  idx -= U * U;
  if (idx < U * KH) { int n = idx / KH, k = idx - n * KH; Wh1T[idx] = (_Float16)Wh1[k * U + n]; return; }
  idx -= U * KH;
  if (idx < U) { WinfF[idx] = Winf[idx]; return; }
  idx -= U;
  if (idx < U) { Wx3F[idx] = Wx3[idx]; return; }
}

// Fold Wh2 @ Wout -> Wh2oT[n64][k128], and bh2o = bh2 @ Wout + bout
__global__ void prep_wh2o(const float* __restrict__ Wh2,
                          const float* __restrict__ Wout,
                          const float* __restrict__ bh2,
                          const float* __restrict__ bout, _Float16* Wh2oT,
                          float* bh2o) {
  int idx = blockIdx.x * blockDim.x + threadIdx.x;
  if (idx < HD * U) {
    int n = idx >> 7, k = idx & 127;
    float acc = 0.0f;
#pragma unroll 8
    for (int t = 0; t < U; ++t) acc += Wh2[k * U + t] * Wout[t * HD + n];
    Wh2oT[idx] = (_Float16)acc;
  }
  if (idx < HD) {
    float acc = bout[idx];
#pragma unroll 8
    for (int t = 0; t < U; ++t) acc += bh2[t] * Wout[t * HD + idx];
    bh2o[idx] = acc;
  }
}

// ---------------- main fused edge kernel ----------------
// grid = Bn*Nn workgroups (one per (b,i)); block = 256 (8 waves).
// Wave w owns output columns [16w, 16w+16).
__global__ __launch_bounds__(256) void egnn_edge_kernel(
    const float* __restrict__ x, const float* __restrict__ h,
    const float* __restrict__ be1, const float* __restrict__ be2,
    const float* __restrict__ binf, const float* __restrict__ bx1,
    const float* __restrict__ bx2, const float* __restrict__ bx3,
    const _Float16* __restrict__ We1T, const _Float16* __restrict__ We2T,
    const _Float16* __restrict__ Wx1T, const _Float16* __restrict__ Wx2T,
    const float* __restrict__ WinfF, const float* __restrict__ Wx3F,
    float* __restrict__ mi_ws, float* __restrict__ xout) {
  const int wg = blockIdx.x;
  const int b = wg >> 9;
  const int i = wg & (Nn - 1);
  const int tid = threadIdx.x;
  const int lane = tid & 31;
  const int wave = tid >> 5;
  const int ncol = (wave << 4) | (lane & 15);
  const int mhalf = (lane & 16) ? 8 : 0;
  const int frow = tid >> 4;       // 0..15, fill-phase row
  const int fcol0 = tid & 15;      // fill-phase column start

  __shared__ __align__(16) _Float16 featBuf[16 * KE];  // aliased as buf2[16][U]
  __shared__ __align__(16) _Float16 buf1[16 * U];
  __shared__ float diffBuf[16][3];
  __shared__ float edot[16];
  __shared__ float pdot[16];
  __shared__ _Float16 hiS[HD];
  __shared__ float xiS[3];

  if (tid < HD) hiS[tid] = (_Float16)h[(b * Nn + i) * HD + tid];
  if (tid < 3) xiS[tid] = x[(b * Nn + i) * 3 + tid];
  if (tid < 16) { edot[tid] = 0.0f; pdot[tid] = 0.0f; }

  // per-lane constants for this wave's column slab
  const float be1v = be1[ncol], be2v = be2[ncol];
  const float bx1v = bx1[ncol], bx2v = bx2[ncol];
  const float winfv = WinfF[ncol], wx3v = Wx3F[ncol];
  const float binfs = binf[0], bx3s = bx3[0];

  // preload all B fragments into registers (reused for all 32 j-tiles)
  v16h bWe1[5], bWe2[4], bWx1[4], bWx2[4];
#pragma unroll
  for (int s = 0; s < 5; ++s) bWe1[s] = glbB(We1T, KE, wave << 4, 32 * s, lane);
#pragma unroll
  for (int s = 0; s < 4; ++s) bWe2[s] = glbB(We2T, U, wave << 4, 32 * s, lane);
#pragma unroll
  for (int s = 0; s < 4; ++s) bWx1[s] = glbB(Wx1T, U, wave << 4, 32 * s, lane);
#pragma unroll
  for (int s = 0; s < 4; ++s) bWx2[s] = glbB(Wx2T, U, wave << 4, 32 * s, lane);

  float accMi[8] = {0, 0, 0, 0, 0, 0, 0, 0};
  float sacc0 = 0.0f, sacc1 = 0.0f, sacc2 = 0.0f;  // wave0 lanes<16 only

  __syncthreads();

  for (int jt = 0; jt < Nn / 16; ++jt) {
    const int jbase = jt << 4;

    // ---- build feature tile: [norm | h_j | h_i | 0-pad] ----
    if (tid < 16) {
      int j = jbase + tid;
      float dx0 = x[(b * Nn + j) * 3 + 0] - xiS[0];
      float dx1 = x[(b * Nn + j) * 3 + 1] - xiS[1];
      float dx2 = x[(b * Nn + j) * 3 + 2] - xiS[2];
      if (j == i) { dx0 = 0.0f; dx1 = 0.0f; dx2 = 0.0f; }
      diffBuf[tid][0] = dx0; diffBuf[tid][1] = dx1; diffBuf[tid][2] = dx2;
      featBuf[tid * KE] = (_Float16)sqrtf(dx0 * dx0 + dx1 * dx1 + dx2 * dx2);
    }
    {
      // thread t fills row tid>>4, columns (tid&15) + 16k  (division-free)
      const float* __restrict__ hj = h + (b * Nn + jbase + frow) * HD - 1;
      _Float16* __restrict__ dst = featBuf + frow * KE;
#pragma unroll
      for (int k = 0; k < KE / 16; ++k) {
        int col = fcol0 + 16 * k;
        if (col == 0) continue;  // norm slot, written above
        float v;
        if (col <= HD)          v = hj[col];
        else if (col <= 2 * HD) v = (float)hiS[col - HD - 1];
        else                    v = 0.0f;
        dst[col] = (_Float16)v;
      }
    }
    __syncthreads();  // B0: feat ready; edot/pdot zeroed

    // ---- GEMM1: elu(feat @ We1 + be1) -> buf1 ----
    {
      v16h a[5];
#pragma unroll
      for (int s = 0; s < 5; ++s) a[s] = ldsA(featBuf, KE, 32 * s, lane);
      v8f c1 = {};
#pragma unroll
      for (int s = 0; s < 5; ++s) c1 = wmma32(a[s], bWe1[s], c1);
#pragma unroll
      for (int r = 0; r < 8; ++r)
        buf1[(mhalf + r) * U + ncol] = (_Float16)eluf(c1[r] + be1v);
    }
    __syncthreads();  // B1

    // ---- GEMM2: m_ij = (buf1 @ We2 + be2) * mask -> regs + buf2(LDS) ----
    float m32[8];
    {
      v16h a[4];
#pragma unroll
      for (int s = 0; s < 4; ++s) a[s] = ldsA(buf1, U, 32 * s, lane);
      v8f c2 = {};
#pragma unroll
      for (int s = 0; s < 4; ++s) c2 = wmma32(a[s], bWe2[s], c2);
#pragma unroll
      for (int r = 0; r < 8; ++r) {
        int j = jbase + mhalf + r;
        float z = (j == i) ? 0.0f : (c2[r] + be2v);
        m32[r] = z;
        featBuf[(mhalf + r) * U + ncol] = (_Float16)z;  // buf2 (alias, ld=U)
      }
    }
    // e-dot partial sums: per-row sum over this wave's 16 columns, then LDS add
#pragma unroll
    for (int r = 0; r < 8; ++r) {
      float sv = m32[r] * winfv;
      sv += __shfl_xor(sv, 1, 32);
      sv += __shfl_xor(sv, 2, 32);
      sv += __shfl_xor(sv, 4, 32);
      sv += __shfl_xor(sv, 8, 32);
      if ((lane & 15) == 0) atomicAdd(&edot[mhalf + r], sv);
    }
    __syncthreads();  // B2: buf2 + edot complete

    // ---- m_i accumulation: accMi += m_ij * sigmoid(edot + binf) ----
#pragma unroll
    for (int r = 0; r < 8; ++r) {
      int M = mhalf + r;
      int j = jbase + M;
      float ev = (j == i) ? 0.0f : sigf(edot[M] + binfs);
      accMi[r] += m32[r] * ev;
    }

    // ---- GEMM3: hid1 = elu(m_ij @ Wx1 + bx1) -> buf1 (overwrite) ----
    {
      v16h a[4];
#pragma unroll
      for (int s = 0; s < 4; ++s) a[s] = ldsA(featBuf, U, 32 * s, lane);
      v8f c3 = {};
#pragma unroll
      for (int s = 0; s < 4; ++s) c3 = wmma32(a[s], bWx1[s], c3);
#pragma unroll
      for (int r = 0; r < 8; ++r)
        buf1[(mhalf + r) * U + ncol] = (_Float16)eluf(c3[r] + bx1v);
    }
    __syncthreads();  // B3
    if (wave == 1 && lane < 16) edot[lane] = 0.0f;  // re-arm for next tile

    // ---- GEMM4: hid2 = elu(hid1 @ Wx2 + bx2); pdot = hid2 . Wx3 ----
    {
      v16h a[4];
#pragma unroll
      for (int s = 0; s < 4; ++s) a[s] = ldsA(buf1, U, 32 * s, lane);
      v8f c4 = {};
#pragma unroll
      for (int s = 0; s < 4; ++s) c4 = wmma32(a[s], bWx2[s], c4);
#pragma unroll
      for (int r = 0; r < 8; ++r) {
        float sv = eluf(c4[r] + bx2v) * wx3v;
        sv += __shfl_xor(sv, 1, 32);
        sv += __shfl_xor(sv, 2, 32);
        sv += __shfl_xor(sv, 4, 32);
        sv += __shfl_xor(sv, 8, 32);
        if ((lane & 15) == 0) atomicAdd(&pdot[mhalf + r], sv);
      }
    }
    __syncthreads();  // B4: pdot complete

    // ---- shift accumulation: shift += diff * phi (wave0, program-ordered) ----
    if (tid < 16) {
      int j = jbase + tid;
      float phi = (j == i) ? 0.0f : (pdot[tid] + bx3s);
      sacc0 += diffBuf[tid][0] * phi;
      sacc1 += diffBuf[tid][1] * phi;
      sacc2 += diffBuf[tid][2] * phi;
      pdot[tid] = 0.0f;  // re-arm (same threads that read it)
    }
  }

  // ---- write m_i: sum accMi over 16 rows ----
  {
    float t = 0.0f;
#pragma unroll
    for (int r = 0; r < 8; ++r) t += accMi[r];
    t += __shfl_xor(t, 16, 32);  // combine the two row-halves
    if (lane < 16) mi_ws[(b * Nn + i) * U + ncol] = t;
  }

  // ---- reduce shift across 16 lanes, write x_out ----
  if (tid < 16) {
    sacc0 += __shfl_xor(sacc0, 1, 32); sacc0 += __shfl_xor(sacc0, 2, 32);
    sacc0 += __shfl_xor(sacc0, 4, 32); sacc0 += __shfl_xor(sacc0, 8, 32);
    sacc1 += __shfl_xor(sacc1, 1, 32); sacc1 += __shfl_xor(sacc1, 2, 32);
    sacc1 += __shfl_xor(sacc1, 4, 32); sacc1 += __shfl_xor(sacc1, 8, 32);
    sacc2 += __shfl_xor(sacc2, 1, 32); sacc2 += __shfl_xor(sacc2, 2, 32);
    sacc2 += __shfl_xor(sacc2, 4, 32); sacc2 += __shfl_xor(sacc2, 8, 32);
    if (tid == 0) {
      const float inv = 1.0f / (float)(Nn - 1);
      xout[(b * Nn + i) * 3 + 0] = xiS[0] + sacc0 * inv;
      xout[(b * Nn + i) * 3 + 1] = xiS[1] + sacc1 * inv;
      xout[(b * Nn + i) * 3 + 2] = xiS[2] + sacc2 * inv;
    }
  }
}

// ---------------- node kernel: h_new ----------------
// grid = Bn*Nn/16 workgroups; block = 256 (8 waves).
__global__ __launch_bounds__(256) void egnn_node_kernel(
    const float* __restrict__ h, const float* __restrict__ mi_ws,
    const float* __restrict__ bh1, const _Float16* __restrict__ Wh1T,
    const _Float16* __restrict__ Wh2oT, const float* __restrict__ bh2o,
    float* __restrict__ hout) {
  const int gbase = blockIdx.x * 16;
  const int tid = threadIdx.x;
  const int lane = tid & 31;
  const int wave = tid >> 5;
  const int ncol = (wave << 4) | (lane & 15);
  const int mhalf = (lane & 16) ? 8 : 0;

  __shared__ __align__(16) _Float16 feat2[16 * KH];
  __shared__ __align__(16) _Float16 bufh[16 * U];

  {
    // division-free fill: row = tid>>4, cols = (tid&15) + 16k
    const int frow = tid >> 4;
    const int g = gbase + frow;  // g == b*Nn + i
    _Float16* __restrict__ dst = feat2 + frow * KH;
#pragma unroll
    for (int k = 0; k < KH / 16; ++k) {
      int col = (tid & 15) + 16 * k;
      float v = (col < U) ? mi_ws[g * U + col] : h[g * HD + (col - U)];
      dst[col] = (_Float16)v;
    }
  }

  const float bh1v = bh1[ncol];
  v16h bW1[6];
#pragma unroll
  for (int s = 0; s < 6; ++s) bW1[s] = glbB(Wh1T, KH, wave << 4, 32 * s, lane);
  __syncthreads();

  {
    v16h a[6];
#pragma unroll
    for (int s = 0; s < 6; ++s) a[s] = ldsA(feat2, KH, 32 * s, lane);
    v8f c = {};
#pragma unroll
    for (int s = 0; s < 6; ++s) c = wmma32(a[s], bW1[s], c);
#pragma unroll
    for (int r = 0; r < 8; ++r)
      bufh[(mhalf + r) * U + ncol] = (_Float16)eluf(c[r] + bh1v);
  }
  __syncthreads();

  if (wave < 4) {  // only 64 output columns
    v16h bW2[4];
#pragma unroll
    for (int s = 0; s < 4; ++s)
      bW2[s] = glbB(Wh2oT, U, wave << 4, 32 * s, lane);
    const float b2 = bh2o[ncol];
    v16h a[4];
#pragma unroll
    for (int s = 0; s < 4; ++s) a[s] = ldsA(bufh, U, 32 * s, lane);
    v8f c2 = {};
#pragma unroll
    for (int s = 0; s < 4; ++s) c2 = wmma32(a[s], bW2[s], c2);
#pragma unroll
    for (int r = 0; r < 8; ++r) {
      int g = gbase + mhalf + r;
      hout[g * HD + ncol] = c2[r] + b2;
    }
  }
}

// ---------------- launch ----------------
extern "C" void kernel_launch(void* const* d_in, const int* in_sizes, int n_in,
                              void* d_out, int out_size, void* d_ws,
                              size_t ws_size, hipStream_t stream) {
  (void)in_sizes; (void)n_in; (void)out_size; (void)ws_size;
  const float* x    = (const float*)d_in[0];
  const float* h    = (const float*)d_in[1];
  const float* We1  = (const float*)d_in[2];
  const float* be1  = (const float*)d_in[3];
  const float* We2  = (const float*)d_in[4];
  const float* be2  = (const float*)d_in[5];
  const float* Winf = (const float*)d_in[6];
  const float* binf = (const float*)d_in[7];
  const float* Wx1  = (const float*)d_in[8];
  const float* bx1  = (const float*)d_in[9];
  const float* Wx2  = (const float*)d_in[10];
  const float* bx2  = (const float*)d_in[11];
  const float* Wx3  = (const float*)d_in[12];
  const float* bx3  = (const float*)d_in[13];
  const float* Wh1  = (const float*)d_in[14];
  const float* bh1  = (const float*)d_in[15];
  const float* Wh2  = (const float*)d_in[16];
  const float* bh2  = (const float*)d_in[17];
  const float* Wout = (const float*)d_in[18];
  const float* bout = (const float*)d_in[19];

  char* ws = (char*)d_ws;
  float*     mi_ws = (float*)(ws + OFF_MI);
  _Float16*  We1T  = (_Float16*)(ws + OFF_WE1T);
  _Float16*  We2T  = (_Float16*)(ws + OFF_WE2T);
  _Float16*  Wx1T  = (_Float16*)(ws + OFF_WX1T);
  _Float16*  Wx2T  = (_Float16*)(ws + OFF_WX2T);
  _Float16*  Wh1T  = (_Float16*)(ws + OFF_WH1T);
  _Float16*  Wh2oT = (_Float16*)(ws + OFF_WH2OT);
  float*     WinfF = (float*)(ws + OFF_WINF);
  float*     Wx3F  = (float*)(ws + OFF_WX3);
  float*     bh2o  = (float*)(ws + OFF_BH2O);

  float* xout = (float*)d_out;                 // [Bn*Nn*3]
  float* hout = xout + Bn * Nn * 3;            // [Bn*Nn*HD]

  const int packTotal = U * KE + 3 * U * U + U * KH + 2 * U;
  prep_pack<<<(packTotal + 255) / 256, 256, 0, stream>>>(
      We1, We2, Wx1, Wx2, Wh1, Winf, Wx3, We1T, We2T, Wx1T, Wx2T, Wh1T, WinfF,
      Wx3F);
  prep_wh2o<<<(HD * U + 255) / 256, 256, 0, stream>>>(Wh2, Wout, bh2, bout,
                                                      Wh2oT, bh2o);
  egnn_edge_kernel<<<Bn * Nn, 256, 0, stream>>>(
      x, h, be1, be2, binf, bx1, bx2, bx3, We1T, We2T, Wx1T, Wx2T, WinfF, Wx3F,
      mi_ws, xout);
  egnn_node_kernel<<<Bn * Nn / 16, 256, 0, stream>>>(h, mi_ws, bh1, Wh1T,
                                                     Wh2oT, bh2o, hout);
}
